// GNConvolution_76733885710815
// MI455X (gfx1250) — matrix operand order repaired and checked by reading, not compiled, not run on hardware
//
#include <hip/hip_runtime.h>
#include <hip/hip_bf16.h>
#include <math.h>

// ---------------------------------------------------------------------------
// MEGNet-style edge convolution for MI455X (gfx1250), wave32 + WMMA bf16.
// ---------------------------------------------------------------------------

#define N_NODES   50000
#define N_EDGES   1600000
#define ATOM_D    64
#define EDGE_D    32
#define STATE_D   32
#define CONCAT_D  192            // 2*64 + 32 + 32
#define OUT_D     64

#define BLOCK_THREADS    256     // 8 waves (wave32)
#define EDGES_PER_BLOCK  128     // 16 edges per wave
#define FSTRIDE          200     // feats row stride in bf16 elems (192 + pad, even; 400B row)

#define NFRAG            24      // 6 K-steps * 4 N-tiles per matrix
#define FRAG_DWORDS      (32 * 8)            // 256 dwords (1KB) per fragment
#define WMAT_DWORDS      (2 * NFRAG * FRAG_DWORDS)   // = 12288 dwords per matrix pair slot
#define WFRAG_DWORDS     (2 * NFRAG * FRAG_DWORDS)   // S + G = 12288 dwords total... see layout
// Layout: wfrag[mat(2)][frag(24)][lane(32)][j(8)]  -> 2*24*256 = 12288 dwords = 49152 B
#define WFRAG_TOTAL_DW   (2 * NFRAG * FRAG_DWORDS)
#define SMEM_BYTES       (WFRAG_TOTAL_DW * 4 + EDGES_PER_BLOCK * FSTRIDE * 2)

typedef __attribute__((ext_vector_type(16))) __bf16 v16bf;
typedef __attribute__((ext_vector_type(8)))  float  v8f;

union Frag {
    v16bf        v;
    unsigned int u[8];
    uint4        q[2];
};

__device__ __forceinline__ unsigned short f2bf(float f) {
    union { float f; unsigned int u; } x; x.f = f;
    unsigned int r = x.u + 0x7FFFu + ((x.u >> 16) & 1u);   // round-to-nearest-even
    return (unsigned short)(r >> 16);
}

// Pack two f32 -> one dword of two bf16 (packed hw convert when available).
__device__ __forceinline__ unsigned int pack2bf(float lo, float hi) {
#if __has_builtin(__builtin_amdgcn_cvt_pk_bf16_f32)
    typedef __attribute__((ext_vector_type(2))) __bf16 v2bf;
    union { v2bf v; unsigned int u; } r;
    r.v = __builtin_amdgcn_cvt_pk_bf16_f32(lo, hi);
    return r.u;
#else
    return (unsigned int)f2bf(lo) | ((unsigned int)f2bf(hi) << 16);
#endif
}

__device__ __forceinline__ uint2 pack4bf(float4 f) {
    uint2 r;
    r.x = pack2bf(f.x, f.y);
    r.y = pack2bf(f.z, f.w);
    return r;
}

__device__ __forceinline__ float softplusf(float x) {
    return (x > 20.f) ? x : log1pf(__expf(x));
}

// ---------------------------------------------------------------------------
// Kernel 1: out = atom_features   (out doubles as the segment-sum accumulator)
// ---------------------------------------------------------------------------
__global__ __launch_bounds__(BLOCK_THREADS)
void gnc_init_out(const float4* __restrict__ atom4, float4* __restrict__ out4, int n4) {
    int i = blockIdx.x * blockDim.x + threadIdx.x;
    if (i < n4) out4[i] = atom4[i];
}

// ---------------------------------------------------------------------------
// Kernel 2: per-edge gather -> bf16 WMMA GEMMs -> gated message -> atomic scatter
// ---------------------------------------------------------------------------
__global__ __launch_bounds__(BLOCK_THREADS)
void gnc_edge_kernel(const float* __restrict__ atom,
                     const float* __restrict__ bond,
                     const float* __restrict__ state,
                     const float* __restrict__ kernel_s,
                     const float* __restrict__ bias_s,
                     const float* __restrict__ kernel_g,
                     const float* __restrict__ bias_g,
                     const int*   __restrict__ pair,
                     const int*   __restrict__ bond_graph,
                     float*       __restrict__ out) {
    extern __shared__ unsigned int smem[];
    // wfrag[mat][frag][lane][j]: B operands pre-swizzled into the exact WMMA
    // per-lane VGPR layout, so hot-loop fills are ds_load_b128 w/ immediate offsets.
    unsigned int*   wfrag = smem;                                    // 12288 dwords
    unsigned short* feats = (unsigned short*)(smem + WFRAG_TOTAL_DW);// [128][FSTRIDE]

    const int t    = threadIdx.x;
    const int lane = t & 31;
    const int wave = t >> 5;

    // ---- stage weights into fragment order -------------------------------
    // B layout (16x16x32 bf16): lane L holds column n = nt*16 + (L&15);
    // dword j holds rows k = kk*32 + (L>>4)*16 + 2j, +1.
    #pragma unroll 2
    for (int mat = 0; mat < 2; ++mat) {
        const float* W = mat ? kernel_g : kernel_s;
        unsigned int* dstf = wfrag + mat * (NFRAG * FRAG_DWORDS);
        for (int idx = t; idx < NFRAG * FRAG_DWORDS; idx += BLOCK_THREADS) {
            int j  = idx & 7;
            int L  = (idx >> 3) & 31;
            int f  = idx >> 8;            // 0..23 : f = kk*4 + nt
            int kk = f >> 2, nt = f & 3;
            int n  = nt * 16 + (L & 15);
            int k  = kk * 32 + (L >> 4) * 16 + 2 * j;
            dstf[idx] = pack2bf(W[k * OUT_D + n], W[(k + 1) * OUT_D + n]);
        }
    }

    // ---- stage this wave's 16-edge feature tile (2 lanes cooperate per edge) ----
    const int m    = lane & 15;                         // edge row within wave tile
    const int half = lane >> 4;                         // which half of the 192 cols
    const int edge = blockIdx.x * EDGES_PER_BLOCK + wave * 16 + m;
    const int src  = pair[2 * edge];                    // atom_i / segment target
    const int nbr  = pair[2 * edge + 1];                // atom_j
    const int gidx = bond_graph[edge];

    uint2* row2 = (uint2*)&feats[(wave * 16 + m) * FSTRIDE];   // 8B-aligned (400B rows)
    if (half == 0) {
        const float4* ai = (const float4*)(atom + (size_t)src * ATOM_D);
        const float4* aj = (const float4*)(atom + (size_t)nbr * ATOM_D);
        #pragma unroll
        for (int q = 0; q < 16; ++q) row2[q] = pack4bf(ai[q]);          // cols 0..63
        #pragma unroll
        for (int q = 0; q < 8; ++q)  row2[16 + q] = pack4bf(aj[q]);     // cols 64..95
    } else {
        const float4* aj = (const float4*)(atom  + (size_t)nbr  * ATOM_D);
        const float4* st = (const float4*)(state + (size_t)gidx * STATE_D);
        const float4* bd = (const float4*)(bond  + (size_t)edge * EDGE_D);
        #pragma unroll
        for (int q = 0; q < 8; ++q)  row2[24 + q] = pack4bf(aj[8 + q]); // cols 96..127
        #pragma unroll
        for (int q = 0; q < 8; ++q)  row2[32 + q] = pack4bf(st[q]);     // cols 128..159
        #pragma unroll
        for (int q = 0; q < 8; ++q)  row2[40 + q] = pack4bf(bd[q]);     // cols 160..191
    }
    __syncthreads();

    // ---- GEMM: [16 x 192] x [192 x 64] twice, f32 accumulate ----
    v8f accS[4] = {};                                   // 4 N-tiles of 16
    v8f accG[4] = {};

    const unsigned short* arow = &feats[(wave * 16 + (lane & 15)) * FSTRIDE];
    const int aHi = (lane >> 4) * 8;                    // lane-group K offset for A

    // Per-lane fragment bases: frag f lives at +f*1024B (imm offsets for ds_load_b128)
    const uint4* bS = (const uint4*)wfrag + lane * 2;                // S fragments
    const uint4* bG = bS + NFRAG * (FRAG_DWORDS / 4);                // G fragments (+24KB)

    #pragma unroll
    for (int kk = 0; kk < 6; ++kk) {                    // K = kk*32 .. kk*32+31
        // A fragment: 16x32 bf16, contiguous k = 16*(lane>>4) + s, same order as B (A row striped within a lane)
        Frag a;
        #pragma unroll
        for (int j = 0; j < 8; ++j) {
            int kloc = kk * 32 + 2 * aHi + 2 * j;
            a.u[j] = *(const unsigned int*)&arow[kloc];
        }
        #pragma unroll
        for (int nt = 0; nt < 4; ++nt) {
            const int f = kk * 4 + nt;                  // fragment index
            Frag bs, bg;
            bs.q[0] = bS[f * 64];
            bs.q[1] = bS[f * 64 + 1];
            bg.q[0] = bG[f * 64];
            bg.q[1] = bG[f * 64 + 1];
            accS[nt] = __builtin_amdgcn_wmma_f32_16x16x32_bf16(
                false, a.v, false, bs.v, (short)0, accS[nt], false, false);
            accG[nt] = __builtin_amdgcn_wmma_f32_16x16x32_bf16(
                false, a.v, false, bg.v, (short)0, accG[nt], false, false);
        }
    }

    // ---- epilogue: bias, sigmoid*softplus, atomic scatter to out[dst] ----
    // C/D layout: VGPR r, lanes 0-15 -> M=r, N=lane; lanes 16-31 -> M=8+r, N=lane-16
    const int ncol = lane & 15;
    int dstn[8];
    #pragma unroll
    for (int r = 0; r < 8; ++r) {
        int mm = (lane >> 4) * 8 + r;                   // lanes 0-15 hold src for m=0..15
        dstn[r] = __shfl(src, mm, 32);
    }
    #pragma unroll
    for (int nt = 0; nt < 4; ++nt) {
        const int col = nt * 16 + ncol;
        const float bS_ = bias_s[col];
        const float bG_ = bias_g[col];
        #pragma unroll
        for (int r = 0; r < 8; ++r) {
            float ts  = accS[nt][r] + bS_;
            float tg  = accG[nt][r] + bG_;
            float sig = 1.f / (1.f + __expf(-ts));
            float msg = sig * softplusf(tg);
            atomicAdd(&out[(size_t)dstn[r] * OUT_D + col], msg);
        }
    }
}

// ---------------------------------------------------------------------------
// Kernel 3: out = softplus(out)   (out already holds atom + agg)
// ---------------------------------------------------------------------------
__global__ __launch_bounds__(BLOCK_THREADS)
void gnc_softplus(float* __restrict__ out, int n) {
    int i = blockIdx.x * blockDim.x + threadIdx.x;
    if (i < n) out[i] = softplusf(out[i]);
}

// ---------------------------------------------------------------------------
extern "C" void kernel_launch(void* const* d_in, const int* in_sizes, int n_in,
                              void* d_out, int out_size, void* d_ws, size_t ws_size,
                              hipStream_t stream) {
    (void)in_sizes; (void)n_in; (void)d_ws; (void)ws_size; (void)out_size;

    const float* atom       = (const float*)d_in[0];
    const float* bond       = (const float*)d_in[1];
    const float* state      = (const float*)d_in[2];
    const float* kernel_s   = (const float*)d_in[3];
    const float* bias_s     = (const float*)d_in[4];
    const float* kernel_g   = (const float*)d_in[5];
    const float* bias_g     = (const float*)d_in[6];
    const int*   pair       = (const int*)d_in[7];
    const int*   bond_graph = (const int*)d_in[9];
    float*       out        = (float*)d_out;

    // 1) out = atom (accumulator init), float4-vectorized
    const int n4 = (N_NODES * OUT_D) / 4;
    gnc_init_out<<<(n4 + BLOCK_THREADS - 1) / BLOCK_THREADS, BLOCK_THREADS, 0, stream>>>(
        (const float4*)atom, (float4*)out, n4);

    // 2) edge GEMM + gated message + scatter-add
    const int nblocks = N_EDGES / EDGES_PER_BLOCK;   // 12500 exactly
    gnc_edge_kernel<<<nblocks, BLOCK_THREADS, SMEM_BYTES, stream>>>(
        atom, bond, state, kernel_s, bias_s, kernel_g, bias_g, pair, bond_graph, out);

    // 3) out = softplus(out)
    const int n = N_NODES * OUT_D;
    gnc_softplus<<<(n + BLOCK_THREADS - 1) / BLOCK_THREADS, BLOCK_THREADS, 0, stream>>>(out, n);
}
